// TripleScatterModule_12111807775165
// MI455X (gfx1250) — matrix operand order repaired and checked by reading, not compiled, not run on hardware
//
#include <hip/hip_runtime.h>

#define F_INP 8
#define HIDW  64
#define F_OUT 8
#define R_DIM 256
#define C_DIM 4096
#define NSETS 3
#define N_INDD 4096
#define M_DIM 4096

typedef _Float16 half8 __attribute__((ext_vector_type(8)));
typedef _Float16 v16h  __attribute__((ext_vector_type(16)));
typedef float    v8f   __attribute__((ext_vector_type(8)));

static __device__ __forceinline__ v16h mk16(half8 a, half8 b) {
  v16h r;
#pragma unroll
  for (int e = 0; e < 8; ++e) { r[e] = a[e]; r[e + 8] = b[e]; }
  return r;
}

// Async DMA: global -> LDS, 16B per lane, tracked by ASYNCcnt (CDNA5).
static __device__ __forceinline__ void async_b128(unsigned lds_off,
                                                  unsigned long long gaddr) {
  asm volatile("global_load_async_to_lds_b128 %0, %1, off"
               :: "v"(lds_off), "v"(gaddr) : "memory");
}

// ---------------------------------------------------------------------------
// gcol[s][i][k] = ind_i[s][j][1] where ind_i[s][j][0] == k  (col0 is a perm)
// ---------------------------------------------------------------------------
__global__ void __launch_bounds__(256)
k_buildtab(const int* __restrict__ i0, const int* __restrict__ i1,
           const int* __restrict__ i2, int* __restrict__ gtab)
{
  int t  = blockIdx.x * 256 + threadIdx.x;     // (s*3+i)*N_INDD + j
  int si = t / N_INDD;
  int j  = t - si * N_INDD;
  int s  = si / 3, i = si - 3 * s;
  const int* ip = (i == 0) ? i0 : (i == 1) ? i1 : i2;
  int k = ip[((size_t)s * N_INDD + j) * 2 + 0];
  int v = ip[((size_t)s * N_INDD + j) * 2 + 1];
  gtab[si * N_INDD + k] = v;
}

// ---------------------------------------------------------------------------
// input[f][r][c] (f32) -> T[c][r][f] (f16): per-column block = 4KB contiguous
// ---------------------------------------------------------------------------
__global__ void __launch_bounds__(256)
k_transpose(const float* __restrict__ in, _Float16* __restrict__ T)
{
  int idx = blockIdx.x * 256 + threadIdx.x;    // idx = r*C + c
  int r = idx / C_DIM;
  int c = idx - r * C_DIM;
  half8 h;
#pragma unroll
  for (int f = 0; f < F_INP; ++f)
    h[f] = (_Float16)in[(size_t)f * (R_DIM * C_DIM) + idx];   // coalesced reads
  *(half8*)&T[((size_t)c * R_DIM + r) * F_INP] = h;
}

// ---------------------------------------------------------------------------
// One-time weight prep: padded f16 images of w1^T-ready and w2 layouts (6KB).
// wp[0..2047]    = w1h[n][k pad32]   (k<24 valid)
// wp[2048..3071] = w2h[n pad16][k]   (n<8 valid)
// ---------------------------------------------------------------------------
__global__ void __launch_bounds__(256)
k_prepw(const float* __restrict__ w1, const float* __restrict__ w2,
        _Float16* __restrict__ wp)
{
  int t = threadIdx.x;
  for (int idx = t; idx < HIDW * 32; idx += 256) {
    int k = idx & 31;
    wp[idx] = (k < 24) ? (_Float16)w1[(idx >> 5) * 24 + k] : (_Float16)0.0f;
  }
  for (int idx = t; idx < 16 * HIDW; idx += 256)
    wp[2048 + idx] = (idx < 8 * HIDW) ? (_Float16)w2[idx] : (_Float16)0.0f;
}

// ---------------------------------------------------------------------------
// Main: one block per (set s, column-slot m). 8 waves x 2 row-tiles = 256 rows.
// Layer1: [16x24(pad32)] x [32x64] via 4x wmma_f32_16x16x32_f16
// Layer2: [16x64] x [64x16(8 valid)] via 2x wmma (LDS bounce for lane remap)
// Staging: GLOBAL_LOAD_ASYNC_TO_LDS_B128 + s_wait_asynccnt (ASYNCcnt path)
// Scatter: int atomicMax of relu-clamped f32 (all outputs >= 0 by construction)
// ---------------------------------------------------------------------------
__global__ void __launch_bounds__(256)
k_main(const _Float16* __restrict__ T, const int* __restrict__ gtab,
       const int* __restrict__ mix,
       const int* __restrict__ i0, const int* __restrict__ i1,
       const int* __restrict__ i2,
       const _Float16* __restrict__ wprep,
       const float* __restrict__ b1, const float* __restrict__ b2,
       float* __restrict__ out)
{
  __shared__ __align__(64) _Float16 colbuf[3 * R_DIM * F_INP]; // 12KB [i][r][f]
  __shared__ __align__(64) _Float16 wbuf[3072];                // 6KB: w1h | w2h
  __shared__ __align__(64) _Float16 reluBuf[8 * 16 * HIDW];    // 16KB per-wave [M][HID]
  __shared__ int s_sc[3];
  __shared__ int s_gc[3];

  const int tid = threadIdx.x;
  const int s = blockIdx.x / M_DIM;
  const int m = blockIdx.x - s * M_DIM;

  if (tid < 3) {
    const int* ip = (tid == 0) ? i0 : (tid == 1) ? i1 : i2;
    int mx = mix[(s * 3 + tid) * M_DIM + m];
    s_sc[tid] = ip[((size_t)s * N_INDD + mx) * 2 + 1];   // scatter column
    s_gc[tid] = gtab[(s * 3 + tid) * N_INDD + mx];       // gather column
  }

  // Async-stage the prepared weights (6KB = 384 x 16B).
  {
    unsigned wb0 = (unsigned)(uintptr_t)(&wbuf[0]);
    unsigned long long wg = (unsigned long long)(uintptr_t)wprep;
#pragma unroll
    for (int u = tid; u < 384; u += 256)
      async_b128(wb0 + (unsigned)(u << 4), wg + ((unsigned long long)u << 4));
  }
  __syncthreads();   // publish s_gc/s_sc

  // Async-stage the 3 gathered columns (3 x 4KB, each contiguous in T).
  {
    unsigned cb0 = (unsigned)(uintptr_t)(&colbuf[0]);
    unsigned long long tb = (unsigned long long)(uintptr_t)T;
#pragma unroll
    for (int i = 0; i < 3; ++i) {
      unsigned long long ga =
          tb + ((unsigned long long)(unsigned)s_gc[i] << 12) +
          (unsigned long long)(tid << 4);
      async_b128(cb0 + (unsigned)(i * 4096 + (tid << 4)), ga);
    }
  }
  asm volatile("s_wait_asynccnt 0" ::: "memory");
  __syncthreads();

  const int lane = tid & 31;
  const int wv = tid >> 5;
  const int lo = lane & 15;
  const int hi = lane >> 4;   // 0 or 1

  // Constant B fragments (32x16 f16 layout: lane<16 -> K=h, lane>=16 -> K=16+h)
  v16h bf0 = *(const v16h*)&wbuf[(0 * 16 + lo) * 32 + hi * 16];
  v16h bf1 = *(const v16h*)&wbuf[(1 * 16 + lo) * 32 + hi * 16];
  v16h bf2 = *(const v16h*)&wbuf[(2 * 16 + lo) * 32 + hi * 16];
  v16h bf3 = *(const v16h*)&wbuf[(3 * 16 + lo) * 32 + hi * 16];
  v16h b2f0 = *(const v16h*)&wbuf[2048 + lo * HIDW + 0 * 32 + hi * 16];
  v16h b2f1 = *(const v16h*)&wbuf[2048 + lo * HIDW + 1 * 32 + hi * 16];

  float bias1[4];
#pragma unroll
  for (int nt = 0; nt < 4; ++nt) bias1[nt] = b1[nt * 16 + lo];
  const float bias2 = (lo < 8) ? b2[lo] : 0.0f;

  _Float16* wrelu = &reluBuf[wv * 16 * HIDW];
  const half8 zero8 = {};

#pragma unroll
  for (int tt = 0; tt < 2; ++tt) {
    const int rbase = (wv * 2 + tt) * 16;
    const int rloc = rbase + lo;           // A-matrix row M = lane%16

    // A1 fragment (16x32 f16): K-chunks of 8 = feature blocks of the 3 columns
    half8 u0 = *(const half8*)&colbuf[hi * (2 * R_DIM * F_INP) + rloc * F_INP];
    half8 u1 = hi ? zero8
                  : *(const half8*)&colbuf[1 * (R_DIM * F_INP) + rloc * F_INP];
    v16h a1 = mk16(u0, u1);

    v8f acc0, acc1, acc2, acc3;
#pragma unroll
    for (int e = 0; e < 8; ++e) {
      acc0[e] = bias1[0]; acc1[e] = bias1[1];
      acc2[e] = bias1[2]; acc3[e] = bias1[3];
    }
    acc0 = __builtin_amdgcn_wmma_f32_16x16x32_f16(false, a1, false, bf0, (short)0, acc0, false, false);
    acc1 = __builtin_amdgcn_wmma_f32_16x16x32_f16(false, a1, false, bf1, (short)0, acc1, false, false);
    acc2 = __builtin_amdgcn_wmma_f32_16x16x32_f16(false, a1, false, bf2, (short)0, acc2, false, false);
    acc3 = __builtin_amdgcn_wmma_f32_16x16x32_f16(false, a1, false, bf3, (short)0, acc3, false, false);

    // ReLU -> f16 -> per-wave LDS [M][HID] (lane remap for layer-2 A fragment)
    const int mrow = 8 * hi;
#pragma unroll
    for (int j = 0; j < 8; ++j) {
      int base = (mrow + j) * HIDW;
      wrelu[base + 0 * 16 + lo] = (_Float16)fmaxf(acc0[j], 0.0f);
      wrelu[base + 1 * 16 + lo] = (_Float16)fmaxf(acc1[j], 0.0f);
      wrelu[base + 2 * 16 + lo] = (_Float16)fmaxf(acc2[j], 0.0f);
      wrelu[base + 3 * 16 + lo] = (_Float16)fmaxf(acc3[j], 0.0f);
    }
    asm volatile("" ::: "memory");  // LDS is in-order per wave; pin compiler order

    // Layer 2: two K=32 chunks over HID=64
    v8f d2;
#pragma unroll
    for (int e = 0; e < 8; ++e) d2[e] = bias2;
    {
      half8 r0 = *(const half8*)&wrelu[lo * HIDW + 0 + hi * 16];
      half8 r1 = *(const half8*)&wrelu[lo * HIDW + 8 + hi * 16];
      v16h a2 = mk16(r0, r1);
      d2 = __builtin_amdgcn_wmma_f32_16x16x32_f16(false, a2, false, b2f0, (short)0, d2, false, false);
    }
    {
      half8 r0 = *(const half8*)&wrelu[lo * HIDW + 32 + hi * 16];
      half8 r1 = *(const half8*)&wrelu[lo * HIDW + 40 + hi * 16];
      v16h a2 = mk16(r0, r1);
      d2 = __builtin_amdgcn_wmma_f32_16x16x32_f16(false, a2, false, b2f1, (short)0, d2, false, false);
    }

    // Scatter-max into out[f][r][c] for the 3 target columns.
    // Output baseline is 0 (include_self), so clamp at 0 and use signed-int
    // atomicMax (valid ordering for non-negative IEEE floats).
    if (lo < 8) {
      const int f = lo;
      int c0 = s_sc[0], c1 = s_sc[1], c2 = s_sc[2];
      int* fbase = (int*)out + (size_t)f * (R_DIM * C_DIM);
#pragma unroll
      for (int j = 0; j < 8; ++j) {
        int r = rbase + mrow + j;
        int iv = __float_as_int(fmaxf(d2[j], 0.0f));
        int* rowp = fbase + (size_t)r * C_DIM;
        atomicMax(rowp + c0, iv);
        atomicMax(rowp + c1, iv);
        atomicMax(rowp + c2, iv);
      }
    }
  }
}

extern "C" void kernel_launch(void* const* d_in, const int* in_sizes, int n_in,
                              void* d_out, int out_size, void* d_ws, size_t ws_size,
                              hipStream_t stream)
{
  (void)in_sizes; (void)n_in; (void)ws_size;
  const float* input = (const float*)d_in[0];
  const float* w1 = (const float*)d_in[1];
  const float* b1 = (const float*)d_in[2];
  const float* w2 = (const float*)d_in[3];
  const float* b2 = (const float*)d_in[4];
  const int* i0  = (const int*)d_in[5];
  const int* i1  = (const int*)d_in[6];
  const int* i2  = (const int*)d_in[7];
  const int* mix = (const int*)d_in[8];
  float* out = (float*)d_out;

  char* ws = (char*)d_ws;
  _Float16* T = (_Float16*)ws;                      // 16 MB: [C][R][F] f16
  size_t tbytes = (size_t)C_DIM * R_DIM * F_INP * sizeof(_Float16);
  int* gtab = (int*)(ws + tbytes);                  // 9*4096 ints
  _Float16* wprep = (_Float16*)(ws + tbytes + (size_t)NSETS * 3 * N_INDD * sizeof(int));

  hipMemsetAsync(d_out, 0, (size_t)out_size * sizeof(float), stream);
  k_buildtab<<<(NSETS * 3 * N_INDD) / 256, 256, 0, stream>>>(i0, i1, i2, gtab);
  k_prepw<<<1, 256, 0, stream>>>(w1, w2, wprep);
  k_transpose<<<(R_DIM * C_DIM) / 256, 256, 0, stream>>>(input, T);
  k_main<<<NSETS * M_DIM, 256, 0, stream>>>(T, gtab, mix, i0, i1, i2,
                                            wprep, b1, b2, out);
}
